// CausalTransformer_41034117546202
// MI455X (gfx1250) — matrix operand order repaired and checked by reading, not compiled, not run on hardware
//
#include <hip/hip_runtime.h>
#include <hip/hip_bf16.h>
#include <math.h>

// ---------------------------------------------------------------------------
// CDNA5 / gfx1250 causal-transformer forward.
// All GEMMs + attention einsums run on v_wmma_f32_16x16x32_f16 (wave32).
// GEMM is register-blocked 32x64 per wave: 8 wmma per k-step fed by 12 b128
// loads (1.5 loads/wmma) to cut L2 traffic ~2.7x vs a 16x16-per-wave tile.
// ---------------------------------------------------------------------------

typedef _Float16 f16;
typedef f16   v16h __attribute__((ext_vector_type(16)));
typedef f16   v8h  __attribute__((ext_vector_type(8)));
typedef float v8f  __attribute__((ext_vector_type(8)));

#define LL   6
#define EE   256
#define HH   4
#define DHH  64
#define MM   512
#define NBB  32
#define TT   2048
#define BB   2
#define BT   4096      // B*T
#define OBSD 637
#define OBSP 640       // padded K for obs projection
#define ACTP 64        // padded N for action head (N tile = 64)

// Build a v16h A/B fragment from two contiguous 8-half (16B) chunks.
__device__ __forceinline__ v16h ld16(const f16* p0, const f16* p1) {
  v8h a = *(const v8h*)p0;
  v8h b = *(const v8h*)p1;
  v16h r;
#pragma unroll
  for (int i = 0; i < 8; ++i) { r[i] = a[i]; r[i + 8] = b[i]; }
  return r;
}

__device__ __forceinline__ v8f wmma_acc(v16h a, v16h b, v8f c) {
  return __builtin_amdgcn_wmma_f32_16x16x32_f16(false, a, false, b,
                                                (short)0, c, false, false);
}

// ---------------------------------------------------------------------------
// Register-blocked WMMA GEMM: C[m,n] = A[m,:K] * Bt[n,:K] (Bt pre-transposed).
// One wave computes a 32x64 tile: acc[2][4], 8 wmma per 32-wide k-step.
// A-fragment lane layout (16-bit, 16x32): lanes<16: e0-7->K0-7, e8-15->K16-23;
// lanes>=16: e0-7->K8-15, e8-15->K24-31. B-fragment: lane->N, same K order
// as A. Optional bias / residual / relu; writes fp32
// and/or f16 and/or f16-transposed (for V).
// ---------------------------------------------------------------------------
__global__ void __launch_bounds__(32)
gemm16(const f16* __restrict__ A, const f16* __restrict__ Bt,
       const float* __restrict__ bias, const float* __restrict__ resid,
       float* __restrict__ Cf, f16* __restrict__ Ch, f16* __restrict__ ChT,
       int N, int K, int Nstore, int relu)
{
  const int lane = threadIdx.x;
  const int half = lane >> 4;
  const int l15  = lane & 15;
  const int m0 = blockIdx.y * 32;
  const int n0 = blockIdx.x * 64;

  v8f acc[2][4];
#pragma unroll
  for (int mt = 0; mt < 2; ++mt)
#pragma unroll
    for (int nt = 0; nt < 4; ++nt)
#pragma unroll
      for (int i = 0; i < 8; ++i) acc[mt][nt][i] = 0.f;

  const f16* arow0 = A  + (size_t)(m0 + l15)      * K + half * 8;
  const f16* arow1 = A  + (size_t)(m0 + 16 + l15) * K + half * 8;
  const f16* brow  = Bt + (size_t)(n0 + l15)      * K + half * 8;

  for (int k0 = 0; k0 < K; k0 += 32) {
    v16h a0 = ld16(arow0 + k0, arow0 + k0 + 16);
    v16h a1 = ld16(arow1 + k0, arow1 + k0 + 16);
    // speculative prefetch one 128-col block ahead (dropped if OOB)
    __builtin_prefetch(arow0 + k0 + 128, 0, 0);
    __builtin_prefetch(brow  + k0 + 128, 0, 0);
#pragma unroll
    for (int nt = 0; nt < 4; ++nt) {
      const f16* bp = brow + (size_t)nt * 16 * K + k0;
      v16h b = ld16(bp, bp + 16);
      acc[0][nt] = wmma_acc(a0, b, acc[0][nt]);
      acc[1][nt] = wmma_acc(a1, b, acc[1][nt]);
    }
  }

#pragma unroll
  for (int mt = 0; mt < 2; ++mt)
#pragma unroll
    for (int nt = 0; nt < 4; ++nt)
#pragma unroll
      for (int r = 0; r < 8; ++r) {
        const int m = m0 + mt * 16 + r + half * 8;  // lanes>=16 hold M = 8+r
        const int n = n0 + nt * 16 + l15;
        if (n < Nstore) {
          float v = acc[mt][nt][r];
          if (bias)  v += bias[n];
          if (resid) v += resid[(size_t)m * Nstore + n];
          if (relu)  v = fmaxf(v, 0.f);
          if (Cf)  Cf [(size_t)m * Nstore + n] = v;
          if (Ch)  Ch [(size_t)m * Nstore + n] = (f16)v;
          if (ChT) ChT[(size_t)n * BT + m]     = (f16)v;   // [N][B*T] transposed
        }
      }
}

// ---------------------------------------------------------------------------
// Flash attention: one wave per (b, h, 16-row q tile). Streams k in 32-column
// steps: S = Q*K^T via 4 WMMA, streaming softmax with rel-pos bias from an
// LDS table, P routed through LDS to A-fragment layout, O += P*V via 4 WMMA.
// ---------------------------------------------------------------------------
__global__ void __launch_bounds__(32)
attn16(const f16* __restrict__ q16, const f16* __restrict__ k16,
       const f16* __restrict__ vt16, f16* __restrict__ ao16,
       const float* __restrict__ btab, int l)
{
  __shared__ float sbias[TT];
  __shared__ __align__(16) f16 pbuf[16 * 32];

  const int lane = threadIdx.x;
  const int half = lane >> 4;
  const int l15  = lane & 15;
  const int qt = blockIdx.x;
  const int h  = blockIdx.y;
  const int b  = blockIdx.z;

  for (int i = lane; i < TT; i += 32)
    sbias[i] = btab[((size_t)l * HH + h) * TT + i];
  __syncthreads();

  // Q fragments for d-blocks [0,32) and [32,64)
  const f16* qrow = q16 + ((size_t)(b * TT + qt * 16 + l15)) * EE + h * DHH + half * 8;
  const v16h aQ0 = ld16(qrow,      qrow + 16);
  const v16h aQ1 = ld16(qrow + 32, qrow + 48);

  v8f accO[4];
#pragma unroll
  for (int c = 0; c < 4; ++c)
#pragma unroll
    for (int i = 0; i < 8; ++i) accO[c][i] = 0.f;

  float mrow[8], lrow[8];
#pragma unroll
  for (int r = 0; r < 8; ++r) { mrow[r] = -1e30f; lrow[r] = 0.f; }

  const float scale = 0.125f;                 // DH^-0.5
  const int nkt = (qt * 16 + 15) / 32 + 1;    // causal: k tiles up to diagonal

  for (int kt = 0; kt < nkt; ++kt) {
    const int k0 = kt * 32;

    v8f s0, s1;
#pragma unroll
    for (int i = 0; i < 8; ++i) { s0[i] = 0.f; s1[i] = 0.f; }

#pragma unroll
    for (int dblk = 0; dblk < 64; dblk += 32) {
      const v16h aq = dblk ? aQ1 : aQ0;
      const f16* ka = k16 + ((size_t)(b * TT + k0 + l15)) * EE + h * DHH + dblk + half * 8;
      v16h bka = ld16(ka, ka + 16);
      s0 = wmma_acc(aq, bka, s0);
      const f16* kb = k16 + ((size_t)(b * TT + k0 + 16 + l15)) * EE + h * DHH + dblk + half * 8;
      v16h bkb = ld16(kb, kb + 16);
      s1 = wmma_acc(aq, bkb, s1);
    }

    // streaming softmax (rows live across a 16-lane half; reduce with shfl_xor)
#pragma unroll
    for (int r = 0; r < 8; ++r) {
      const int mloc = r + half * 8;
      const int m = qt * 16 + mloc;
      const int ra = m - (k0 + l15);
      const int rb = ra - 16;
      float sa = (ra >= 0) ? s0[r] * scale + sbias[ra] : -1e30f;
      float sb = (rb >= 0) ? s1[r] * scale + sbias[rb] : -1e30f;
      float mx = fmaxf(sa, sb);
#pragma unroll
      for (int msk = 8; msk >= 1; msk >>= 1)
        mx = fmaxf(mx, __shfl_xor(mx, msk, 32));
      const float mnew  = fmaxf(mrow[r], mx);
      const float alpha = __expf(mrow[r] - mnew);
      const float pa = __expf(sa - mnew);
      const float pb = __expf(sb - mnew);
      float rs = pa + pb;
#pragma unroll
      for (int msk = 8; msk >= 1; msk >>= 1)
        rs += __shfl_xor(rs, msk, 32);
      lrow[r] = lrow[r] * alpha + rs;
      mrow[r] = mnew;
#pragma unroll
      for (int c = 0; c < 4; ++c) accO[c][r] *= alpha;
      pbuf[mloc * 32 + l15]      = (f16)pa;
      pbuf[mloc * 32 + 16 + l15] = (f16)pb;
    }
    __syncthreads();

    // P in A-fragment layout from LDS; V^T fragments contiguous in k
    const f16* pp = pbuf + l15 * 32 + half * 8;
    const v16h aP = ld16(pp, pp + 16);
#pragma unroll
    for (int c = 0; c < 4; ++c) {
      const f16* pv = vt16 + ((size_t)(h * DHH + c * 16 + l15)) * BT
                           + b * TT + k0 + half * 8;
      v16h bv = ld16(pv, pv + 16);
      accO[c] = wmma_acc(aP, bv, accO[c]);
    }
    __syncthreads();
  }

#pragma unroll
  for (int c = 0; c < 4; ++c)
#pragma unroll
    for (int r = 0; r < 8; ++r) {
      const int m = qt * 16 + r + half * 8;
      const int d = h * DHH + c * 16 + l15;
      ao16[((size_t)(b * TT + m)) * EE + d] = (f16)(accO[c][r] / lrow[r]);
    }
}

// ---------------------------------------------------------------------------
// RMSNorm: one 256-thread block per row of E=256; outputs f16 for GEMM input.
// ---------------------------------------------------------------------------
__global__ void __launch_bounds__(256)
rmsnorm_k(const float* __restrict__ x, const float* __restrict__ w,
          f16* __restrict__ out)
{
  __shared__ float red[256];
  const int row = blockIdx.x, t = threadIdx.x;
  const float v = x[(size_t)row * EE + t];
  red[t] = v * v;
  __syncthreads();
  for (int s = 128; s > 0; s >>= 1) {
    if (t < s) red[t] += red[t + s];
    __syncthreads();
  }
  const float inv = rsqrtf(red[0] / (float)EE + 1.1920929e-07f);
  out[(size_t)row * EE + t] = (f16)(v * inv * w[t]);
}

// bias table: btab[l][h][rel] = relb[l][bucket(rel)][h], rel in [0,T)
__global__ void biastab_k(const float* __restrict__ relb, float* __restrict__ btab)
{
  const int idx = blockIdx.x * blockDim.x + threadIdx.x;
  if (idx >= LL * HH * TT) return;
  const int rel = idx % TT;
  const int h   = (idx / TT) % HH;
  const int l   = idx / (TT * HH);
  int bucket;
  if (rel < 8) {
    bucket = rel;
  } else {
    int lg = (int)(logf((float)rel * 0.125f) * (8.0f / logf(16.0f)));
    bucket = 8 + (lg < 7 ? lg : 7);
  }
  if (rel > 0) bucket += 16;   // bidirectional split (positive side)
  btab[idx] = relb[((size_t)l * NBB + bucket) * HH + h];
}

// fp32 [K,N] weight -> f16 [Npad,Kpad] transposed, zero padded
__global__ void convw_k(const float* __restrict__ src, f16* __restrict__ dst,
                        int K, int N, int Kpad, int Npad)
{
  const int idx = blockIdx.x * blockDim.x + threadIdx.x;
  if (idx >= Npad * Kpad) return;
  const int n = idx / Kpad, k = idx % Kpad;
  dst[idx] = (k < K && n < N) ? (f16)src[(size_t)k * N + n] : (f16)0.f;
}

// obs fp32 [BT,637] -> f16 [BT,640] zero padded
__global__ void convobs_k(const float* __restrict__ src, f16* __restrict__ dst)
{
  const int idx = blockIdx.x * blockDim.x + threadIdx.x;
  if (idx >= BT * OBSP) return;
  const int r = idx / OBSP, c = idx % OBSP;
  dst[idx] = (c < OBSD) ? (f16)src[(size_t)r * OBSD + c] : (f16)0.f;
}

// ---------------------------------------------------------------------------
extern "C" void kernel_launch(void* const* d_in, const int* in_sizes, int n_in,
                              void* d_out, int out_size, void* d_ws, size_t ws_size,
                              hipStream_t stream)
{
  (void)in_sizes; (void)n_in; (void)out_size; (void)ws_size;

  const float* obs     = (const float*)d_in[0];
  const float* obs_w   = (const float*)d_in[1];
  const float* obs_b   = (const float*)d_in[2];
  const float* norm1_w = (const float*)d_in[3];
  const float* wq      = (const float*)d_in[4];
  const float* wk      = (const float*)d_in[5];
  const float* wv      = (const float*)d_in[6];
  const float* wo      = (const float*)d_in[7];
  const float* relb    = (const float*)d_in[8];
  const float* norm2_w = (const float*)d_in[9];
  const float* w1      = (const float*)d_in[10];
  const float* b1      = (const float*)d_in[11];
  const float* w2      = (const float*)d_in[12];
  const float* b2      = (const float*)d_in[13];
  const float* fn_w    = (const float*)d_in[14];
  const float* act_w   = (const float*)d_in[15];
  const float* act_b   = (const float*)d_in[16];
  const float* obsh_w  = (const float*)d_in[17];
  const float* obsh_b  = (const float*)d_in[18];
  float* dout = (float*)d_out;

  // workspace bump allocator (~33 MB total)
  char* wp = (char*)d_ws;
  auto alloc = [&](size_t bytes) -> void* {
    void* r = (void*)wp;
    wp += (bytes + 255) & ~(size_t)255;
    return r;
  };
  f16* wqT    = (f16*)alloc((size_t)LL * EE * EE * 2);
  f16* wkT    = (f16*)alloc((size_t)LL * EE * EE * 2);
  f16* wvT    = (f16*)alloc((size_t)LL * EE * EE * 2);
  f16* woT    = (f16*)alloc((size_t)LL * EE * EE * 2);
  f16* w1T    = (f16*)alloc((size_t)LL * EE * MM * 2);
  f16* w2T    = (f16*)alloc((size_t)LL * MM * EE * 2);
  f16* obswT  = (f16*)alloc((size_t)EE * OBSP * 2);
  f16* actwT  = (f16*)alloc((size_t)ACTP * EE * 2);
  f16* obshwT = (f16*)alloc((size_t)OBSP * EE * 2);
  f16* obs16  = (f16*)alloc((size_t)BT * OBSP * 2);
  f16* h16    = (f16*)alloc((size_t)BT * EE * 2);
  f16* q16    = (f16*)alloc((size_t)BT * EE * 2);
  f16* kk16   = (f16*)alloc((size_t)BT * EE * 2);
  f16* vt16   = (f16*)alloc((size_t)EE * BT * 2);   // transposed V
  f16* ao16   = (f16*)alloc((size_t)BT * EE * 2);
  f16* ffn16  = (f16*)alloc((size_t)BT * MM * 2);
  f16* hf16   = (f16*)alloc((size_t)BT * EE * 2);
  float* x    = (float*)alloc((size_t)BT * EE * 4);
  float* btab = (float*)alloc((size_t)LL * HH * TT * 4);

  const dim3 blk256(256);
  auto cv = [&](const float* s, f16* d, int K, int N, int Kp, int Np) {
    convw_k<<<dim3((Np * Kp + 255) / 256), blk256, 0, stream>>>(s, d, K, N, Kp, Np);
  };

  // --- one-time conversions (deterministic every call) ---
  convobs_k<<<dim3((BT * OBSP + 255) / 256), blk256, 0, stream>>>(obs, obs16);
  for (int l = 0; l < LL; ++l) {
    cv(wq + (size_t)l * EE * EE, wqT + (size_t)l * EE * EE, EE, EE, EE, EE);
    cv(wk + (size_t)l * EE * EE, wkT + (size_t)l * EE * EE, EE, EE, EE, EE);
    cv(wv + (size_t)l * EE * EE, wvT + (size_t)l * EE * EE, EE, EE, EE, EE);
    cv(wo + (size_t)l * EE * EE, woT + (size_t)l * EE * EE, EE, EE, EE, EE);
    cv(w1 + (size_t)l * EE * MM, w1T + (size_t)l * EE * MM, EE, MM, EE, MM);
    cv(w2 + (size_t)l * MM * EE, w2T + (size_t)l * MM * EE, MM, EE, MM, EE);
  }
  cv(obs_w,  obswT,  OBSD, EE, OBSP, EE);
  cv(act_w,  actwT,  EE, 4,    EE, ACTP);
  cv(obsh_w, obshwT, EE, OBSD, EE, OBSP);
  biastab_k<<<dim3((LL * HH * TT + 255) / 256), blk256, 0, stream>>>(relb, btab);

  const dim3 wave(32);
  const int MB = BT / 32;   // 128 row-tiles of 32
  // x = obs16 @ obs_w + obs_b
  gemm16<<<dim3(EE / 64, MB), wave, 0, stream>>>(
      obs16, obswT, obs_b, nullptr, x, nullptr, nullptr, EE, OBSP, EE, 0);

  for (int l = 0; l < LL; ++l) {
    const size_t wo4 = (size_t)l * EE * EE;
    const size_t wf  = (size_t)l * EE * MM;
    rmsnorm_k<<<dim3(BT), blk256, 0, stream>>>(x, norm1_w + l * EE, h16);
    gemm16<<<dim3(EE / 64, MB), wave, 0, stream>>>(
        h16, wqT + wo4, nullptr, nullptr, nullptr, q16, nullptr, EE, EE, EE, 0);
    gemm16<<<dim3(EE / 64, MB), wave, 0, stream>>>(
        h16, wkT + wo4, nullptr, nullptr, nullptr, kk16, nullptr, EE, EE, EE, 0);
    gemm16<<<dim3(EE / 64, MB), wave, 0, stream>>>(
        h16, wvT + wo4, nullptr, nullptr, nullptr, nullptr, vt16, EE, EE, EE, 0);
    attn16<<<dim3(TT / 16, HH, BB), wave, 0, stream>>>(q16, kk16, vt16, ao16, btab, l);
    gemm16<<<dim3(EE / 64, MB), wave, 0, stream>>>(      // x += attn_out @ wo
        ao16, woT + wo4, nullptr, x, x, nullptr, nullptr, EE, EE, EE, 0);
    rmsnorm_k<<<dim3(BT), blk256, 0, stream>>>(x, norm2_w + l * EE, h16);
    gemm16<<<dim3(MM / 64, MB), wave, 0, stream>>>(      // relu(h @ w1 + b1)
        h16, w1T + wf, b1 + l * MM, nullptr, nullptr, ffn16, nullptr, MM, EE, MM, 1);
    gemm16<<<dim3(EE / 64, MB), wave, 0, stream>>>(      // x += ffn @ w2 + b2
        ffn16, w2T + wf, b2 + l * EE, x, x, nullptr, nullptr, EE, MM, EE, 0);
  }

  rmsnorm_k<<<dim3(BT), blk256, 0, stream>>>(x, fn_w, hf16);
  gemm16<<<dim3(ACTP / 64, MB), wave, 0, stream>>>(      // action logits
      hf16, actwT, act_b, nullptr, dout, nullptr, nullptr, ACTP, EE, 4, 0);
  gemm16<<<dim3(OBSP / 64, MB), wave, 0, stream>>>(      // obs prediction
      hf16, obshwT, obsh_b, nullptr, dout + (size_t)BT * 4, nullptr, nullptr,
      OBSP, EE, OBSD, 0);
}
